// RingAttentionSimulated_30322469110504
// MI455X (gfx1250) — matrix operand order repaired and checked by reading, not compiled, not run on hardware
//
#include <hip/hip_runtime.h>

// ---------------------------------------------------------------------------
// CDNA5 (gfx1250, wave32) causal MHA: bf16 WMMA GEMMs + flash-style attention
// ---------------------------------------------------------------------------

typedef __bf16 bf16_t;
typedef __attribute__((ext_vector_type(16))) __bf16 v16bf;
typedef __attribute__((ext_vector_type(8)))  __bf16 v8bf;
typedef __attribute__((ext_vector_type(4)))  __bf16 v4bf;
typedef __attribute__((ext_vector_type(8)))  float  v8f;

#define B_SZ  2
#define T_SZ  2048
#define D_SZ  1024
#define NH_SZ 16
#define DH_SZ 64
#define M_SZ  (B_SZ * T_SZ)   // 4096 rows

__device__ __forceinline__ v16bf cat8(v8bf lo, v8bf hi) {
  return __builtin_shufflevector(lo, hi, 0, 1, 2, 3, 4, 5, 6, 7,
                                         8, 9, 10, 11, 12, 13, 14, 15);
}

__device__ __forceinline__ v8f wmma_bf16(v16bf a, v16bf b, v8f c) {
  // v_wmma_f32_16x16x32_bf16  (8-arg form, probe-confirmed)
  return __builtin_amdgcn_wmma_f32_16x16x32_bf16(
      /*neg_a=*/false, a, /*neg_b=*/false, b,
      /*c_mod=*/(short)0, c, /*reuse_a=*/false, /*reuse_b=*/false);
}

// --------------------------- fp32 -> bf16 ----------------------------------
__global__ __launch_bounds__(256)
void k_cvt_bf16(const float* __restrict__ in, bf16_t* __restrict__ out, int n) {
  int i = (blockIdx.x * blockDim.x + threadIdx.x) * 4;
  if (i + 3 < n) {
    float4 f = *(const float4*)(in + i);
    v4bf o;
    o[0] = (bf16_t)f.x; o[1] = (bf16_t)f.y;
    o[2] = (bf16_t)f.z; o[3] = (bf16_t)f.w;
    *(v4bf*)(out + i) = o;
  }
}

// -------------------- fp32 [K][N] -> bf16 [N][K] (transpose) ---------------
__global__ __launch_bounds__(256)
void k_transpose_bf16(const float* __restrict__ W, bf16_t* __restrict__ Wt) {
  __shared__ float tile[32][33];
  const int x = blockIdx.x * 32 + threadIdx.x;   // n
  const int y0 = blockIdx.y * 32;                // k tile
#pragma unroll
  for (int j = 0; j < 4; ++j) {
    int y = y0 + threadIdx.y + j * 8;
    tile[threadIdx.y + j * 8][threadIdx.x] = W[(size_t)y * D_SZ + x];
  }
  __syncthreads();
  const int xo = y0 + threadIdx.x;               // k
#pragma unroll
  for (int j = 0; j < 4; ++j) {
    int yo = blockIdx.x * 32 + threadIdx.y + j * 8;  // n
    Wt[(size_t)yo * D_SZ + xo] = (bf16_t)tile[threadIdx.x][threadIdx.y + j * 8];
  }
}

// ------------------------------ GEMM ---------------------------------------
// C[M,N] = A[M,K](bf16,row-major) * B[K,N] given as Bt[N,K] (bf16).
// Block tile 64(M) x 128(N), K-step 32; 8 waves, 4 WMMA accumulators each.
// MODE 0: write bf16 head-major  [b][h][t][d]      (Q, K)
// MODE 1: write bf16 transposed  [b][h][d][t]      (V)
// MODE 2: write fp32 row-major   [gm][gn]          (final output)
// DO_SCALE: multiply result by 1/sqrt(DH) (exact pow2) -- used for Q.
template <int MODE, bool DO_SCALE>
__global__ __launch_bounds__(256)
void k_gemm_wmma(const bf16_t* __restrict__ A, const bf16_t* __restrict__ Bt,
                 void* __restrict__ Cout, int M, int N, int K) {
  __shared__ bf16_t As[64][32];    // [m][k]
  __shared__ bf16_t Bs[128][32];   // [n][k]

  const int tid   = threadIdx.x;
  const int lane  = tid & 31;
  const int wave  = tid >> 5;
  const int lrow  = lane & 15;     // matrix row (A) / col (B,C)
  const int lhi   = lane >> 4;     // half-wave select
  const int waveM = wave & 3;      // 4 row tiles of 16
  const int waveN = wave >> 2;     // 2 col halves of 64
  const int blockRow = blockIdx.x * 64;
  const int blockCol = blockIdx.y * 128;

  v8f acc[4] = {};

  // cooperative-load indices: A, 8 elems/thread; B, 16 elems/thread
  const int lar = tid >> 2;              // 0..63
  const int lac = (tid & 3) * 8;         // 0,8,16,24
  const int lbr = tid >> 1;              // 0..127
  const int lbc = (tid & 1) * 16;        // 0,16

  for (int kb = 0; kb < K; kb += 32) {
    *(v8bf*)&As[lar][lac]     = *(const v8bf*)&A [(size_t)(blockRow + lar) * K + kb + lac];
    *(v8bf*)&Bs[lbr][lbc]     = *(const v8bf*)&Bt[(size_t)(blockCol + lbr) * K + kb + lbc];
    *(v8bf*)&Bs[lbr][lbc + 8] = *(const v8bf*)&Bt[(size_t)(blockCol + lbr) * K + kb + lbc + 8];
    __syncthreads();

    // A-operand: row = waveM*16+lrow; K = 16*lhi .. +15 contiguous
    const bf16_t* ap = &As[waveM * 16 + lrow][16 * lhi];
    v16bf a = cat8(*(const v8bf*)ap, *(const v8bf*)(ap + 8));
#pragma unroll
    for (int j = 0; j < 4; ++j) {
      // B-operand: col = waveN*64 + j*16 + lrow; K = 16*lhi .. +15 contiguous
      const bf16_t* bp = &Bs[waveN * 64 + j * 16 + lrow][16 * lhi];
      v16bf b = cat8(*(const v8bf*)bp, *(const v8bf*)(bp + 8));
      acc[j] = wmma_bf16(a, b, acc[j]);
    }
    __syncthreads();
  }

  // C layout: lane holds column lrow; VGPR r = row (r + 8*lhi)
#pragma unroll
  for (int j = 0; j < 4; ++j) {
    const int gn = blockCol + waveN * 64 + j * 16 + lrow;
#pragma unroll
    for (int r = 0; r < 8; ++r) {
      const int gm = blockRow + waveM * 16 + r + 8 * lhi;
      float v = acc[j][r];
      if (DO_SCALE) v *= 0.125f;          // 1/sqrt(64), exact in bf16
      if (MODE == 2) {
        ((float*)Cout)[(size_t)gm * N + gn] = v;
      } else {
        const int b = gm >> 11, t = gm & (T_SZ - 1);
        const int h = gn >> 6,  d = gn & (DH_SZ - 1);
        if (MODE == 0)
          ((bf16_t*)Cout)[((size_t)(b * NH_SZ + h) * T_SZ + t) * DH_SZ + d] = (bf16_t)v;
        else
          ((bf16_t*)Cout)[((size_t)(b * NH_SZ + h) * DH_SZ + d) * T_SZ + t] = (bf16_t)v;
      }
    }
  }
}

// --------------------- causal flash attention ------------------------------
// One wave per 16-row query tile. Q (pre-scaled)/K: [b][h][t][d] bf16;
// V: [b][h][d][t] bf16. Output Ob: [b][t][h*64+d] bf16 (merged heads).
__global__ __launch_bounds__(32)
void k_flash_attn(const bf16_t* __restrict__ Qh, const bf16_t* __restrict__ Kh,
                  const bf16_t* __restrict__ Vt, bf16_t* __restrict__ Ob) {
  __shared__ bf16_t Pl[16][32];

  const int lane = threadIdx.x;
  const int lrow = lane & 15;
  const int lhi  = lane >> 4;
  const int bh   = blockIdx.y;          // b*NH + h
  const int t0   = blockIdx.x * 16;

  const bf16_t* Qp = Qh + (size_t)bh * T_SZ * DH_SZ;
  const bf16_t* Kp = Kh + (size_t)bh * T_SZ * DH_SZ;
  const bf16_t* Vp = Vt + (size_t)bh * DH_SZ * T_SZ;

  // Q as A-operand, resident for the whole key loop (d split 0..31 / 32..63)
  v16bf qa[2];
#pragma unroll
  for (int kk = 0; kk < 2; ++kk) {
    const bf16_t* qp = Qp + (size_t)(t0 + lrow) * DH_SZ + kk * 32 + 16 * lhi;
    qa[kk] = cat8(*(const v8bf*)qp, *(const v8bf*)(qp + 8));
  }

  float mrow[8], lsum[8];
  v8f oacc[4] = {};
#pragma unroll
  for (int r = 0; r < 8; ++r) { mrow[r] = -1e30f; lsum[r] = 0.f; }

  // one 32-key chunk; MASK selects the causal-diagonal path
  auto do_chunk = [&](int kbase, bool mask) {
    // S = Q * K^T  (two 16x16 halves, contraction 64 = 2 WMMAs each)
    v8f s[2];
#pragma unroll
    for (int j = 0; j < 2; ++j) {
      const bf16_t* kp = Kp + (size_t)(kbase + j * 16 + lrow) * DH_SZ + 16 * lhi;
      v16bf b0 = cat8(*(const v8bf*)kp,        *(const v8bf*)(kp + 8));
      v16bf b1 = cat8(*(const v8bf*)(kp + 32), *(const v8bf*)(kp + 40));
      v8f z = {};
      z = wmma_bf16(qa[0], b0, z);
      z = wmma_bf16(qa[1], b1, z);
      s[j] = z;
    }

    float mloc[8];
#pragma unroll
    for (int r = 0; r < 8; ++r) {
      float v0 = s[0][r];
      float v1 = s[1][r];
      if (mask) {
        const int qrow = t0 + r + 8 * lhi;
        if (kbase + lrow > qrow)      v0 = -1e30f;
        if (kbase + 16 + lrow > qrow) v1 = -1e30f;
        s[0][r] = v0; s[1][r] = v1;
      }
      mloc[r] = fmaxf(v0, v1);
    }
#pragma unroll
    for (int off = 1; off < 16; off <<= 1)
#pragma unroll
      for (int r = 0; r < 8; ++r)
        mloc[r] = fmaxf(mloc[r], __shfl_xor(mloc[r], off, 32));

    // online softmax update; stash P (bf16) to LDS for C-layout -> A-layout
    float psum[8];
#pragma unroll
    for (int r = 0; r < 8; ++r) {
      const float mn = fmaxf(mrow[r], mloc[r]);
      const float sc = __expf(mrow[r] - mn);
      mrow[r] = mn;
      const float p0 = __expf(s[0][r] - mn);
      const float p1 = __expf(s[1][r] - mn);
      Pl[r + 8 * lhi][lrow]      = (bf16_t)p0;
      Pl[r + 8 * lhi][16 + lrow] = (bf16_t)p1;
      psum[r] = p0 + p1;
      lsum[r] *= sc;
#pragma unroll
      for (int n = 0; n < 4; ++n) oacc[n][r] *= sc;
    }
#pragma unroll
    for (int off = 1; off < 16; off <<= 1)
#pragma unroll
      for (int r = 0; r < 8; ++r)
        psum[r] += __shfl_xor(psum[r], off, 32);
#pragma unroll
    for (int r = 0; r < 8; ++r) lsum[r] += psum[r];

    __syncthreads();

    // P as A-operand; V^T gives contiguous B-operand; O += P * V
    const bf16_t* pp = &Pl[lrow][16 * lhi];
    v16bf pa = cat8(*(const v8bf*)pp, *(const v8bf*)(pp + 8));
#pragma unroll
    for (int n = 0; n < 4; ++n) {
      const bf16_t* vp = Vp + (size_t)(n * 16 + lrow) * T_SZ + kbase + 16 * lhi;
      v16bf vb = cat8(*(const v8bf*)vp, *(const v8bf*)(vp + 8));
      oacc[n] = wmma_bf16(pa, vb, oacc[n]);
    }
    __syncthreads();
  };

  const int nch   = (t0 + 16 + 31) >> 5;                      // total chunks
  const int nfull = (t0 >= 31) ? (((t0 - 31) >> 5) + 1) : 0;  // kbase+31 <= t0

  for (int c = 0; c < nfull; ++c) do_chunk(c * 32, false);    // unmasked bulk
  for (int c = nfull; c < nch; ++c) do_chunk(c * 32, true);   // diagonal

  const int b = bh >> 4, h = bh & (NH_SZ - 1);
#pragma unroll
  for (int n = 0; n < 4; ++n)
#pragma unroll
    for (int r = 0; r < 8; ++r) {
      const int trow = t0 + r + 8 * lhi;
      const float v = oacc[n][r] / lsum[r];
      Ob[((size_t)(b * T_SZ + trow)) * D_SZ + h * DH_SZ + n * 16 + lrow] = (bf16_t)v;
    }
}

// ---------------------------------------------------------------------------
extern "C" void kernel_launch(void* const* d_in, const int* in_sizes, int n_in,
                              void* d_out, int out_size, void* d_ws, size_t ws_size,
                              hipStream_t stream) {
  (void)in_sizes; (void)n_in; (void)out_size; (void)ws_size;

  const float* x  = (const float*)d_in[0];
  const float* Wq = (const float*)d_in[1];
  const float* Wk = (const float*)d_in[2];
  const float* Wv = (const float*)d_in[3];
  const float* Wo = (const float*)d_in[4];

  char* ws = (char*)d_ws;
  size_t off = 0;
  auto alloc = [&](size_t bytes) -> char* {
    char* p = ws + off;
    off += (bytes + 255) & ~(size_t)255;
    return p;
  };

  bf16_t* xb  = (bf16_t*)alloc((size_t)M_SZ * D_SZ * sizeof(bf16_t));  // 8 MB
  bf16_t* WqT = (bf16_t*)alloc((size_t)D_SZ * D_SZ * sizeof(bf16_t));  // 2 MB
  bf16_t* WkT = (bf16_t*)alloc((size_t)D_SZ * D_SZ * sizeof(bf16_t));
  bf16_t* WvT = (bf16_t*)alloc((size_t)D_SZ * D_SZ * sizeof(bf16_t));
  bf16_t* WoT = (bf16_t*)alloc((size_t)D_SZ * D_SZ * sizeof(bf16_t));
  bf16_t* Qh  = (bf16_t*)alloc((size_t)M_SZ * D_SZ * sizeof(bf16_t));  // [b][h][t][d], pre-scaled
  bf16_t* Kh  = (bf16_t*)alloc((size_t)M_SZ * D_SZ * sizeof(bf16_t));  // [b][h][t][d]
  bf16_t* Vt  = (bf16_t*)alloc((size_t)M_SZ * D_SZ * sizeof(bf16_t));  // [b][h][d][t]
  bf16_t* Obf = (bf16_t*)alloc((size_t)M_SZ * D_SZ * sizeof(bf16_t));  // [b][t][h*64+d]

  // 1) precision conversion + weight transposes
  k_cvt_bf16<<<(M_SZ * D_SZ / 4 + 255) / 256, 256, 0, stream>>>(x, xb, M_SZ * D_SZ);
  dim3 tb(32, 8), tg(D_SZ / 32, D_SZ / 32);
  k_transpose_bf16<<<tg, tb, 0, stream>>>(Wq, WqT);
  k_transpose_bf16<<<tg, tb, 0, stream>>>(Wk, WkT);
  k_transpose_bf16<<<tg, tb, 0, stream>>>(Wv, WvT);
  k_transpose_bf16<<<tg, tb, 0, stream>>>(Wo, WoT);

  // 2) QKV projections (WMMA GEMMs); Q gets the 1/sqrt(DH) scale folded in
  dim3 gg(M_SZ / 64, D_SZ / 128);
  k_gemm_wmma<0, true ><<<gg, 256, 0, stream>>>(xb, WqT, Qh, M_SZ, D_SZ, D_SZ);
  k_gemm_wmma<0, false><<<gg, 256, 0, stream>>>(xb, WkT, Kh, M_SZ, D_SZ, D_SZ);
  k_gemm_wmma<1, false><<<gg, 256, 0, stream>>>(xb, WvT, Vt, M_SZ, D_SZ, D_SZ);

  // 3) causal flash attention
  dim3 ag(T_SZ / 16, B_SZ * NH_SZ);
  k_flash_attn<<<ag, 32, 0, stream>>>(Qh, Kh, Vt, Obf);

  // 4) output projection -> fp32
  k_gemm_wmma<2, false><<<gg, 256, 0, stream>>>(Obf, WoT, d_out, M_SZ, D_SZ, D_SZ);
}
